// MultiScaleAdaptiveGCN_70600672412048
// MI455X (gfx1250) — compile-verified, matrix operand order checked by reading, not run on hardware
//
#include <hip/hip_runtime.h>
#include <stdint.h>

typedef __attribute__((ext_vector_type(16))) __bf16 v16bf;
typedef __attribute__((ext_vector_type(8)))  float  v8f;

#define B_   64
#define C_   256
#define T_   128
#define V_   25
#define K_   3
#define G_   4
#define CG_  64
#define TT_  16
#define NT_  400   // TT_*V_

// LDS row strides (bf16 units, all even so K-pairs are 4B aligned)
#define XS_   36   // Xbf rows (c*16+t)=1024, cols w 0..31 (pad 25..31 = 0)
#define GS_   68   // Xag rows n=400 (+1 dump), cols c 0..63
#define WSTR_ 66   // Wbf rows o=64 per k, cols c 0..63
#define AS_   32   // Asp rows v (pad to 32), cols w (pad to 32)

// Load a 16x32 bf16 WMMA fragment from LDS stored row-major [row][k], even stride.
// Per ISA 7.12.2: lane<16 holds row=row0+lane, K = {0..7,16..23}; lane>=16 same row,
// K = {8..15,24..31}; pairs packed per VGPR. B uses the mirrored layout ([n][k]).
__device__ __forceinline__ v16bf ldfrag(const __bf16* p, int stride, int row0, int col0, int lane) {
  int half = (lane >> 4) & 1;
  const __bf16* base = p + (size_t)(row0 + (lane & 15)) * stride + col0 + half * 8;
  union { uint32_t u[8]; v16bf v; } f;
#pragma unroll
  for (int i = 0; i < 4; ++i) f.u[i]     = *(const uint32_t*)(base + 2 * i);
#pragma unroll
  for (int i = 0; i < 4; ++i) f.u[4 + i] = *(const uint32_t*)(base + 16 + 2 * i);
  return f.v;
}

// ---- Phase 1a: temporal mean  xm[b][c][v] = mean_t x[b][c][t][v] ----
__global__ __launch_bounds__(256) void gcn_xmean(const float* __restrict__ x, float* __restrict__ xm) {
  int idx = blockIdx.x * 256 + threadIdx.x;
  if (idx >= B_ * C_ * V_) return;
  int v = idx % V_, bc = idx / V_;
  const float* p = x + (size_t)bc * T_ * V_ + v;
  float s = 0.f;
  for (int t = 0; t < T_; ++t) s += p[t * V_];
  xm[idx] = s * (1.0f / T_);
}

// ---- Phase 1b: q = Wq@xm, k = Wk@xm  (B,64,V) each ----
__global__ __launch_bounds__(256) void gcn_qk(const float* __restrict__ Wq, const float* __restrict__ Wk,
                                              const float* __restrict__ xm,
                                              float* __restrict__ qb, float* __restrict__ kb) {
  int idx = blockIdx.x * 256 + threadIdx.x;
  if (idx >= 2 * B_ * 64 * V_) return;
  int which = idx / (B_ * 64 * V_);
  int rem = idx % (B_ * 64 * V_);
  int v = rem % V_;
  int bo = rem / V_;
  int o = bo % 64, b = bo / 64;
  const float* wp = (which ? Wk : Wq) + o * C_;
  const float* xp = xm + (size_t)b * C_ * V_ + v;
  float s = 0.f;
  for (int c = 0; c < C_; ++c) s += wp[c] * xp[c * V_];
  (which ? kb : qb)[rem] = s;
}

// ---- Phase 1c: attention + combine:  acomb[b][k][v][w] = A_norm[k][v][w] + tanh(alpha)*A_dyn[b][v][w]
__global__ __launch_bounds__(256) void gcn_attn(const float* __restrict__ qb, const float* __restrict__ kb,
                                                const float* __restrict__ Aadj, const float* __restrict__ alpha,
                                                float* __restrict__ acomb) {
  __shared__ float qs[G_][16][V_];
  __shared__ float ks[G_][16][V_];
  __shared__ float sc[G_][V_][V_];
  __shared__ float ad[V_][V_];
  __shared__ float rs[K_][V_];
  int b = blockIdx.x, tid = threadIdx.x;
  for (int idx = tid; idx < G_ * 16 * V_; idx += 256) {
    int g = idx / (16 * V_), r = idx % (16 * V_), d = r / V_, v = r % V_;
    qs[g][d][v] = qb[(b * 64 + g * 16 + d) * V_ + v];
    ks[g][d][v] = kb[(b * 64 + g * 16 + d) * V_ + v];
  }
  for (int idx = tid; idx < K_ * V_; idx += 256) {
    int k = idx / V_, v = idx % V_;
    float s = 0.f;
    for (int w = 0; w < V_; ++w) s += Aadj[(k * V_ + v) * V_ + w];
    rs[k][v] = fmaxf(s, 1e-6f);
  }
  __syncthreads();
  for (int idx = tid; idx < G_ * V_ * V_; idx += 256) {
    int g = idx / (V_ * V_), r = idx % (V_ * V_), v = r / V_, w = r % V_;
    float s = 0.f;
    for (int d = 0; d < 16; ++d) s += qs[g][d][v] * ks[g][d][w];
    sc[g][v][w] = s * 0.25f;  // 1/sqrt(dk), dk=16
  }
  __syncthreads();
  if (tid < G_ * V_) {  // row softmax over w
    int g = tid / V_, v = tid % V_;
    float m = -1e30f;
    for (int w = 0; w < V_; ++w) m = fmaxf(m, sc[g][v][w]);
    float ssum = 0.f;
    for (int w = 0; w < V_; ++w) { float e = __expf(sc[g][v][w] - m); sc[g][v][w] = e; ssum += e; }
    float inv = 1.f / ssum;
    for (int w = 0; w < V_; ++w) sc[g][v][w] *= inv;
  }
  __syncthreads();
  for (int idx = tid; idx < V_ * V_; idx += 256) {
    int v = idx / V_, w = idx % V_;
    ad[v][w] = 0.25f * (sc[0][v][w] + sc[1][v][w] + sc[2][v][w] + sc[3][v][w]);
  }
  __syncthreads();
  float al = tanhf(alpha[0]);
  for (int idx = tid; idx < K_ * V_ * V_; idx += 256) {
    int k = idx / (V_ * V_), r = idx % (V_ * V_), v = r / V_, w = r % V_;
    acomb[(size_t)(b * K_ + k) * V_ * V_ + r] = Aadj[(k * V_ + v) * V_ + w] / rs[k][v] + al * ad[v][w];
  }
}

// ---- Phase 2: fused spatial-WMMA + channel-WMMA + BN + residual ----
__global__ __launch_bounds__(256) void gcn_main(const float* __restrict__ x, const float* __restrict__ Wconv,
                                                const float* __restrict__ acomb, const float* __restrict__ gamma,
                                                const float* __restrict__ beta, const float* __restrict__ rmean,
                                                const float* __restrict__ rvar, float* __restrict__ out) {
  __shared__ __align__(16) __bf16 Xbf[CG_ * TT_][XS_];   // [(c,t)][w]  73,728 B
  __shared__ __align__(16) __bf16 Xag[NT_ + 1][GS_];     // [n=(t,v)][c] + dump row; 54,536 B
  __shared__ __align__(16) __bf16 Wbf[K_][CG_][WSTR_];   // 25,344 B
  __shared__ __align__(16) __bf16 Asp[K_][32][AS_];      // B = A_k^T stored [v][w]; 6,144 B
  __shared__ float invS[CG_], shiftS[CG_];

  const int blk = blockIdx.x;
  const int tt = blk & 7, g = (blk >> 3) & 3, b = blk >> 5;
  const int t0 = tt * TT_;
  const int tid = threadIdx.x, lane = tid & 31, wave = tid >> 5;

  // zero pad regions first
  for (int idx = tid; idx < K_ * 32 * AS_; idx += 256) ((__bf16*)Asp)[idx] = (__bf16)0.0f;
  for (int idx = tid; idx < CG_ * TT_ * 7; idx += 256) Xbf[idx / 7][25 + idx % 7] = (__bf16)0.0f;
  __syncthreads();

  if (tid < CG_) {
    int ch = g * CG_ + tid;
    float inv = gamma[ch] * rsqrtf(rvar[ch] + 1e-5f);
    invS[tid] = inv;
    shiftS[tid] = beta[ch] - rmean[ch] * inv;
  }
  for (int idx = tid; idx < K_ * CG_ * CG_; idx += 256) {
    int k = idx / (CG_ * CG_), r = idx % (CG_ * CG_), o = r / CG_, c = r % CG_;
    Wbf[k][o][c] = (__bf16)Wconv[((k * G_ + g) * CG_ + o) * CG_ + c];
  }
  for (int idx = tid; idx < K_ * V_ * V_; idx += 256) {
    int k = idx / (V_ * V_), r = idx % (V_ * V_), v = r / V_, w = r % V_;
    Asp[k][v][w] = (__bf16)acomb[(size_t)(b * K_ + k) * V_ * V_ + r];
  }
  const float* xb = x + (((size_t)b * C_ + g * CG_) * T_ + t0) * V_;
  for (int idx = tid; idx < CG_ * TT_ * V_; idx += 256) {
    int c = idx / (TT_ * V_), r = idx % (TT_ * V_), t = r / V_, w = r % V_;
    Xbf[c * TT_ + t][w] = (__bf16)xb[((size_t)c * T_ + t) * V_ + w];
  }
  __syncthreads();

  // Per-wave fixed roles:
  //  spatial: nt_sp = wave&1 fixed; mt sweeps (wave>>1) + 4i
  //  channel: mt_ch = wave&3 fixed; nt sweeps (wave>>2) + 2s  -> W frags hoisted per k
  const int nt_sp = wave & 1;
  const int mt_ch = wave & 3, half_ch = wave >> 2;
  const int nl = lane & 15, mb = (lane >> 4) * 8;

  v8f facc[13];
  v8f zero = {};
#pragma unroll
  for (int s = 0; s < 13; ++s) facc[s] = zero;

  for (int k = 0; k < K_; ++k) {
    // Spatial: Xag[(t,v)][c] = sum_w Xbf[(c,t)][w] * A_k[v][w]   (M=1024,K=32,N=32)
    {
      v16bf bsp = ldfrag(&Asp[k][0][0], AS_, nt_sp * 16, 0, lane);  // invariant per wave
      const int v = nt_sp * 16 + nl;
      const bool valid = (v < V_);
      // software-pipelined over 16 mtiles: prefetch next A-frag before consuming acc
      v16bf af = ldfrag(&Xbf[0][0], XS_, (wave >> 1) * 16, 0, lane);
#pragma unroll
      for (int i = 0; i < 16; ++i) {
        v8f acc = __builtin_amdgcn_wmma_f32_16x16x32_bf16(false, af, false, bsp, (short)0, zero, false, false);
        v16bf afn = af;
        if (i < 15) afn = ldfrag(&Xbf[0][0], XS_, ((wave >> 1) + 4 * (i + 1)) * 16, 0, lane);
        int mt = (wave >> 1) + 4 * i;
#pragma unroll
        for (int r = 0; r < 8; ++r) {
          int row = mt * 16 + mb + r;                       // row = c*16 + t
          int rn = valid ? ((row & 15) * V_ + v) : NT_;     // invalid lanes -> dump row
          Xag[rn][row >> 4] = (__bf16)acc[r];
        }
        af = afn;
      }
    }
    __syncthreads();
    // Channel: facc += W_k(64x64) @ Xag(64x400); W frags once per k, B frags pipelined
    {
      v16bf wf0 = ldfrag(&Wbf[k][0][0], WSTR_, mt_ch * 16, 0, lane);
      v16bf wf1 = ldfrag(&Wbf[k][0][0], WSTR_, mt_ch * 16, 32, lane);
      v16bf bf0 = ldfrag(&Xag[0][0], GS_, half_ch * 16, 0, lane);
      v16bf bf1 = ldfrag(&Xag[0][0], GS_, half_ch * 16, 32, lane);
#pragma unroll
      for (int s = 0; s < 13; ++s) {
        int nt = half_ch + 2 * s;
        if (nt < 25) {
          v8f acc = facc[s];
          acc = __builtin_amdgcn_wmma_f32_16x16x32_bf16(false, wf0, false, bf0, (short)0, acc, false, false);
          acc = __builtin_amdgcn_wmma_f32_16x16x32_bf16(false, wf1, false, bf1, (short)0, acc, false, false);
          facc[s] = acc;
          int ntn = nt + 2;
          if (ntn < 25) {
            bf0 = ldfrag(&Xag[0][0], GS_, ntn * 16, 0, lane);
            bf1 = ldfrag(&Xag[0][0], GS_, ntn * 16, 32, lane);
          }
        }
      }
    }
    __syncthreads();
  }

  // Epilogue: out = x + BN(sum_k ...)
#pragma unroll
  for (int s = 0; s < 13; ++s) {
    int nt = half_ch + 2 * s;
    if (nt < 25) {
      int n = nt * 16 + nl, t = n / 25, v = n % 25;
#pragma unroll
      for (int r = 0; r < 8; ++r) {
        int o = mt_ch * 16 + mb + r;
        size_t gi = (((size_t)b * C_ + g * CG_ + o) * T_ + t0 + t) * V_ + v;
        out[gi] = x[gi] + facc[s][r] * invS[o] + shiftS[o];
      }
    }
  }
}

extern "C" void kernel_launch(void* const* d_in, const int* in_sizes, int n_in,
                              void* d_out, int out_size, void* d_ws, size_t ws_size,
                              hipStream_t stream) {
  const float* x     = (const float*)d_in[0];
  const float* Aadj  = (const float*)d_in[1];
  const float* Wq    = (const float*)d_in[2];
  const float* Wk    = (const float*)d_in[3];
  const float* Wconv = (const float*)d_in[4];
  const float* alpha = (const float*)d_in[5];
  const float* gamma = (const float*)d_in[6];
  const float* beta  = (const float*)d_in[7];
  const float* rmean = (const float*)d_in[8];
  const float* rvar  = (const float*)d_in[9];
  float* out = (float*)d_out;

  float* ws = (float*)d_ws;
  float* xm = ws;              // 409,600
  float* qb = ws + 409600;     // 102,400
  float* kb = ws + 512000;     // 102,400
  float* ac = ws + 614400;     // 120,000

  gcn_xmean<<<(B_ * C_ * V_ + 255) / 256, 256, 0, stream>>>(x, xm);
  gcn_qk<<<(2 * B_ * 64 * V_ + 255) / 256, 256, 0, stream>>>(Wq, Wk, xm, qb, kb);
  gcn_attn<<<B_, 256, 0, stream>>>(qb, kb, Aadj, alpha, ac);
  gcn_main<<<B_ * G_ * (T_ / TT_), 256, 0, stream>>>(x, Wconv, ac, gamma, beta, rmean, rvar, out);
}
